// Custom_FeedForward_75393855914399
// MI455X (gfx1250) — 8-bit matrix operand order repaired and checked by reading, compiled, not run on hardware
//
#include <hip/hip_runtime.h>
#include <hip/hip_bf16.h>
#include <stdint.h>

typedef __attribute__((ext_vector_type(8))) int v8i;

#define S_DIM 1024
#define E_DIM 4096
#define H_DIM 11008
#define LDS_ROW 80

#define ASYNC_LD16(ldsop, voff, sbase, imm)                                   \
    asm volatile("global_load_async_to_lds_b128 %0, %1, %2 offset:%c3"        \
                 :: "v"((unsigned)(ldsop)), "v"((unsigned)(voff)),            \
                    "s"(sbase), "i"(imm) : "memory")

#define WAIT_ASYNC_LE(n)                                                      \
    asm volatile("s_wait_asynccnt %c0" :: "i"(n) : "memory")

__global__ void pack_i8_kernel(const int* __restrict__ in,
                               signed char* __restrict__ out, long n4) {
    long i = (long)blockIdx.x * blockDim.x + threadIdx.x;
    if (i < n4) {
        int4 v = ((const int4*)in)[i];
        char4 c;
        c.x = (signed char)v.x;
        c.y = (signed char)v.y;
        c.z = (signed char)v.z;
        c.w = (signed char)v.w;
        ((char4*)out)[i] = c;
    }
}

__device__ __forceinline__ float q8f(float v) {
    return fminf(127.0f, fmaxf(-128.0f, rintf(v)));
}

__device__ __forceinline__ v8i load_a_frag(const char* __restrict__ base) {
    uint4 a03 = *(const uint4*)(base + 0);
    uint4 a47 = *(const uint4*)(base + 32);
    v8i a;
    a[0] = (int)a03.x; a[1] = (int)a03.y;
    a[2] = (int)a03.z; a[3] = (int)a03.w;
    a[4] = (int)a47.x; a[5] = (int)a47.y;
    a[6] = (int)a47.z; a[7] = (int)a47.w;
    return a;
}

__device__ __forceinline__ v8i load_b_frag(const char* __restrict__ base) {
    uint4 b03 = *(const uint4*)(base + 0);
    uint4 b47 = *(const uint4*)(base + 32);
    v8i b;
    b[0] = (int)b03.x; b[1] = (int)b03.y; b[2] = (int)b03.z; b[3] = (int)b03.w;
    b[4] = (int)b47.x; b[5] = (int)b47.y; b[6] = (int)b47.z; b[7] = (int)b47.w;
    return b;
}

#define BUFSZ1 25600
__global__ __launch_bounds__(256)
void ffn_gate_up_kernel(const signed char* __restrict__ x8,
                        const signed char* __restrict__ w1,
                        signed char* __restrict__ mulq,
                        const float* __restrict__ scale_x,
                        const float* __restrict__ scale_w1p,
                        const float* __restrict__ scale_w2p,
                        const float* __restrict__ scale_y1,
                        const float* __restrict__ scale_y2,
                        const float* __restrict__ scale_sig,
                        const float* __restrict__ scale_silu_mul,
                        const float* __restrict__ scale_mul) {
    __shared__ char smem[4 * BUFSZ1];

    const int lane = threadIdx.x & 31;
    const int wave = threadIdx.x >> 5;
    const int ln = lane & 15, kh = lane >> 4;
    const int r4 = lane >> 2, c4 = lane & 3;
    const int mw = wave & 3,  nw = wave >> 2;

    const int m0b = blockIdx.y * 64;
    const int n0b = blockIdx.x * 128;

    const unsigned nWdelta = (unsigned)H_DIM * E_DIM;
    const unsigned voffA = (unsigned)(m0b + wave * 8 + r4) * E_DIM + c4 * 16;
    unsigned voffB[4];
#pragma unroll
    for (int j = 0; j < 4; ++j)
        voffB[j] = ((wave < 4) ? 0u : nWdelta) +
                   (unsigned)(n0b + (wave & 3) * 32 + j * 8 + r4) * E_DIM + c4 * 16;

    const int ldsA = (wave * 8 + r4) * LDS_ROW + c4 * 16;
    int ldsB[4];
#pragma unroll
    for (int j = 0; j < 4; ++j)
        ldsB[j] = 5120 + (wave * 32 + j * 8 + r4) * LDS_ROW + c4 * 16;

    v8i acc1[4], acc2[4];
#pragma unroll
    for (int t = 0; t < 4; ++t) {
        acc1[t] = (v8i){0, 0, 0, 0, 0, 0, 0, 0};
        acc2[t] = (v8i){0, 0, 0, 0, 0, 0, 0, 0};
    }

    const signed char* sA = x8;
    const signed char* sB = w1;
    const unsigned smem0 = (unsigned)(uintptr_t)smem;

    ASYNC_LD16(smem0 + ldsA, voffA, sA, 0);
    ASYNC_LD16(smem0 + ldsA + BUFSZ1 - 64, voffA, sA, 64);
#pragma unroll
    for (int j = 0; j < 4; ++j) {
        ASYNC_LD16(smem0 + ldsB[j], voffB[j], sB, 0);
        ASYNC_LD16(smem0 + ldsB[j] + BUFSZ1 - 64, voffB[j], sB, 64);
    }

    const int NK = E_DIM / 64;
    for (int itb = 0; itb < NK; itb += 4) {
#pragma unroll
        for (int j = 0; j < 4; ++j) {
            const int IMM = (j + 2) * 64;
            const int BOF = ((j + 2) & 3) * BUFSZ1 - IMM;
            ASYNC_LD16(smem0 + ldsA + BOF, voffA, sA, IMM);
#pragma unroll
            for (int jj = 0; jj < 4; ++jj)
                ASYNC_LD16(smem0 + ldsB[jj] + BOF, voffB[jj], sB, IMM);

            WAIT_ASYNC_LE(10);
            __syncthreads();

            const char* b = smem + j * BUFSZ1;
            const char* arow = b + (mw * 16 + ln) * LDS_ROW + kh * 16;
            v8i a = load_a_frag(arow);
            const char* brow = b + 5120 + (nw * 64 + ln) * LDS_ROW + kh * 16;
#pragma unroll
            for (int t = 0; t < 4; ++t) {
                v8i bf1 = load_b_frag(brow + t * 16 * LDS_ROW);
                acc1[t] = __builtin_amdgcn_wmma_i32_16x16x64_iu8(
                    true, a, true, bf1, acc1[t], false, false);
                v8i bf2 = load_b_frag(brow + (128 + t * 16) * LDS_ROW);
                acc2[t] = __builtin_amdgcn_wmma_i32_16x16x64_iu8(
                    true, a, true, bf2, acc2[t], false, false);
            }
        }
        sA += 256;
        sB += 256;
    }

    const float sw1 = *scale_w1p;
    const float sw2 = *scale_w2p;
    const int n0w = n0b + nw * 64;

#pragma unroll
    for (int r = 0; r < 8; ++r) {
        const int m = m0b + mw * 16 + r + kh * 8;
        const float sx   = scale_x[m];
        const float sy1  = scale_y1[m];
        const float sy2  = scale_y2[m];
        const float ssig = scale_sig[m];
        const float ssm  = scale_silu_mul[m];
        const float smul = scale_mul[m];
        const float f1 = sx * sw1 / sy1;
        const float f2 = sx * sw2 / sy2;
#pragma unroll
        for (int t = 0; t < 4; ++t) {
            float fc1q = q8f((float)acc1[t][r] * f1);
            float y1   = fc1q * sy1;
            float sig  = 1.0f / (1.0f + __expf(-y1));
            float sigq = q8f(sig / ssig);
            float silq = q8f(y1 * (sigq * ssig) / ssm);
            float fc2q = q8f((float)acc2[t][r] * f2);
            float mq   = q8f((silq * ssm) * (fc2q * sy2) / smul);
            mulq[(size_t)m * H_DIM + n0w + t * 16 + ln] = (signed char)mq;
        }
    }
}

#define BUFSZ2 15360
__global__ __launch_bounds__(256)
void ffn_down_kernel(const signed char* __restrict__ mulq,
                     const signed char* __restrict__ w3,
                     float* __restrict__ out,
                     const float* __restrict__ scale_mul,
                     const float* __restrict__ scale_w3p,
                     const float* __restrict__ scale_y3) {
    __shared__ char smem[4 * BUFSZ2];

    const int lane = threadIdx.x & 31;
    const int wave = threadIdx.x >> 5;
    const int ln = lane & 15, kh = lane >> 4;
    const int r4 = lane >> 2, c4 = lane & 3;
    const int mw = wave & 3,  nw = wave >> 2;

    const int m0b = blockIdx.y * 64;
    const int n0b = blockIdx.x * 128;

    const unsigned voffA = (unsigned)(m0b + wave * 8 + r4) * H_DIM + c4 * 16;
    unsigned voffB[2];
#pragma unroll
    for (int j = 0; j < 2; ++j)
        voffB[j] = (unsigned)(n0b + wave * 16 + j * 8 + r4) * H_DIM + c4 * 16;

    const int ldsA = (wave * 8 + r4) * LDS_ROW + c4 * 16;
    int ldsB[2];
#pragma unroll
    for (int j = 0; j < 2; ++j)
        ldsB[j] = 5120 + (wave * 16 + j * 8 + r4) * LDS_ROW + c4 * 16;

    v8i acc[4];
#pragma unroll
    for (int t = 0; t < 4; ++t) acc[t] = (v8i){0, 0, 0, 0, 0, 0, 0, 0};

    const signed char* sA = mulq;
    const signed char* sB = w3;
    const unsigned smem0 = (unsigned)(uintptr_t)smem;

    ASYNC_LD16(smem0 + ldsA, voffA, sA, 0);
    ASYNC_LD16(smem0 + ldsA + BUFSZ2 - 64, voffA, sA, 64);
#pragma unroll
    for (int j = 0; j < 2; ++j) {
        ASYNC_LD16(smem0 + ldsB[j], voffB[j], sB, 0);
        ASYNC_LD16(smem0 + ldsB[j] + BUFSZ2 - 64, voffB[j], sB, 64);
    }

    const int NK = H_DIM / 64;
    for (int itb = 0; itb < NK; itb += 4) {
#pragma unroll
        for (int j = 0; j < 4; ++j) {
            const int IMM = (j + 2) * 64;
            const int BOF = ((j + 2) & 3) * BUFSZ2 - IMM;
            ASYNC_LD16(smem0 + ldsA + BOF, voffA, sA, IMM);
#pragma unroll
            for (int jj = 0; jj < 2; ++jj)
                ASYNC_LD16(smem0 + ldsB[jj] + BOF, voffB[jj], sB, IMM);

            WAIT_ASYNC_LE(6);
            __syncthreads();

            const char* b = smem + j * BUFSZ2;
            const char* arow = b + (mw * 16 + ln) * LDS_ROW + kh * 16;
            v8i a = load_a_frag(arow);
            const char* brow = b + 5120 + (nw * 64 + ln) * LDS_ROW + kh * 16;
#pragma unroll
            for (int t = 0; t < 4; ++t) {
                v8i bf = load_b_frag(brow + t * 16 * LDS_ROW);
                acc[t] = __builtin_amdgcn_wmma_i32_16x16x64_iu8(
                    true, a, true, bf, acc[t], false, false);
            }
        }
        sA += 256;
        sB += 256;
    }

    const float sw3 = *scale_w3p;
    const int n0w = n0b + nw * 64;
#pragma unroll
    for (int r = 0; r < 8; ++r) {
        const int m = m0b + mw * 16 + r + kh * 8;
        const float f3 = scale_mul[m] * sw3 / scale_y3[m];
#pragma unroll
        for (int t = 0; t < 4; ++t) {
            out[(size_t)m * E_DIM + n0w + t * 16 + ln] = q8f((float)acc[t][r] * f3);
        }
    }
}

__global__ void copy_scale_kernel(const float* __restrict__ s,
                                  float* __restrict__ out_tail, int n) {
    int i = blockIdx.x * blockDim.x + threadIdx.x;
    if (i < n) out_tail[i] = s[i];
}

extern "C" void kernel_launch(void* const* d_in, const int* in_sizes, int n_in,
                              void* d_out, int out_size, void* d_ws, size_t ws_size,
                              hipStream_t stream) {
    const int* x_i32  = (const int*)d_in[0];
    const int* W1_i32 = (const int*)d_in[1];
    const int* W2_i32 = (const int*)d_in[2];
    const int* W3_i32 = (const int*)d_in[3];
    const float* scale_x        = (const float*)d_in[4];
    const float* scale_w1       = (const float*)d_in[5];
    const float* scale_w2       = (const float*)d_in[6];
    const float* scale_w3       = (const float*)d_in[7];
    const float* scale_y1       = (const float*)d_in[8];
    const float* scale_y2       = (const float*)d_in[9];
    const float* scale_sig      = (const float*)d_in[10];
    const float* scale_silu_mul = (const float*)d_in[11];
    const float* scale_mul      = (const float*)d_in[12];
    const float* scale_y3       = (const float*)d_in[13];

    float* out = (float*)d_out;

    const size_t nX = (size_t)S_DIM * E_DIM;
    const size_t nW = (size_t)H_DIM * E_DIM;

    char* ws = (char*)d_ws;
    signed char* x8  = (signed char*)(ws);
    signed char* w18 = (signed char*)(ws + nX);
    signed char* w28 = (signed char*)(ws + nX + nW);
    signed char* w38 = (signed char*)(ws + nX + 2 * nW);
    signed char* mq  = (signed char*)(ws + nX + 3 * nW);
    (void)ws_size; (void)in_sizes; (void)n_in; (void)out_size;

    {
        long n4 = (long)(nX / 4);
        pack_i8_kernel<<<(unsigned)((n4 + 255) / 256), 256, 0, stream>>>(x_i32, x8, n4);
        n4 = (long)(nW / 4);
        pack_i8_kernel<<<(unsigned)((n4 + 255) / 256), 256, 0, stream>>>(W1_i32, w18, n4);
        pack_i8_kernel<<<(unsigned)((n4 + 255) / 256), 256, 0, stream>>>(W2_i32, w28, n4);
        pack_i8_kernel<<<(unsigned)((n4 + 255) / 256), 256, 0, stream>>>(W3_i32, w38, n4);
    }

    {
        dim3 grid(H_DIM / 128, S_DIM / 64);
        ffn_gate_up_kernel<<<grid, 256, 0, stream>>>(
            x8, w18, mq,
            scale_x, scale_w1, scale_w2,
            scale_y1, scale_y2, scale_sig, scale_silu_mul, scale_mul);
    }

    {
        dim3 grid(E_DIM / 128, S_DIM / 64);
        ffn_down_kernel<<<grid, 256, 0, stream>>>(
            mq, w38, out, scale_mul, scale_w3, scale_y3);
    }

    copy_scale_kernel<<<(S_DIM + 255) / 256, 256, 0, stream>>>(
        scale_y3, out + (size_t)S_DIM * E_DIM, S_DIM);
}
